// RawAttentionPivotalTransformer_45440753992155
// MI455X (gfx1250) — matrix operand order repaired and checked by reading, not compiled, not run on hardware
//
#include <hip/hip_runtime.h>

// Problem constants (match reference setup_inputs)
#define BB 2
#define NN 4096
#define CC 512
#define HH 8
#define DDIM 64
#define KP 512
#define C3 1536

typedef __attribute__((ext_vector_type(16))) __bf16 v16bf;
typedef __attribute__((ext_vector_type(8)))  float  v8f;

__device__ __forceinline__ __bf16 f2bf(float f) { return (__bf16)f; }  // native v_cvt

__device__ __forceinline__ v8f wmma_bf16(v16bf a, v16bf b, v8f c) {
  // D = A(16x32) * B(32x16) + C, fp32 accumulate
  return __builtin_amdgcn_wmma_f32_16x16x32_bf16(false, a, false, b, (short)0, c,
                                                 false, false);
}

// ---- fragment loaders (layouts per CDNA5 ISA 7.12.2, wave32) ----
// A (16x32 bf16): lane L<16 -> row L, e[0..7]=K0..7, e[8..15]=K16..23;
//                 lane L>=16 -> row L-16, e[0..7]=K8..15, e[8..15]=K24..31.
// Contiguous: 2x16B per lane.
__device__ __forceinline__ v16bf load_a_bf(const __bf16* A, int ld, int row0,
                                           int k0, int lane) {
  const __bf16* p = A + (size_t)(row0 + (lane & 15)) * ld + k0 + ((lane >> 4) << 3);
  v16bf a;
#pragma unroll
  for (int i = 0; i < 8; ++i) a[i] = p[i];
#pragma unroll
  for (int i = 0; i < 8; ++i) a[8 + i] = p[16 + i];
  return a;
}

// B (32x16) where LDS source is stored transposed (rows = N, cols = K):
// lane holds col n = lane&15, same K order as A: e[0..7]=K(8h..8h+7), e[8..15]=K(16+8h..).
__device__ __forceinline__ v16bf load_bT_bf(const __bf16* Bm, int ld, int n0,
                                            int k0, int lane) {
  const __bf16* p = Bm + (size_t)(n0 + (lane & 15)) * ld + k0 + ((lane >> 4) << 3);
  v16bf b;
#pragma unroll
  for (int i = 0; i < 8; ++i) b[i] = p[i];
#pragma unroll
  for (int i = 0; i < 8; ++i) b[8 + i] = p[16 + i];
  return b;
}

// ---------------- utility kernels ----------------
__global__ __launch_bounds__(256) void cast_bf16_kernel(const float* __restrict__ src,
                                                        __bf16* __restrict__ dst, int n) {
  int i = blockIdx.x * 256 + threadIdx.x;
  if (i < n) dst[i] = f2bf(src[i]);
}

// cast + transpose: src [Kd][Nd] f32 -> dst [Nd][Kd] bf16 (32x32 LDS tiles)
__global__ __launch_bounds__(256) void transpose_cast_kernel(const float* __restrict__ src,
                                                             __bf16* __restrict__ dst,
                                                             int Kd, int Nd) {
  __shared__ float tile[32][33];
  const int n0 = blockIdx.x * 32, k0 = blockIdx.y * 32;
  const int lx = threadIdx.x & 31, ly = threadIdx.x >> 5;  // 32 x 8
#pragma unroll
  for (int i = 0; i < 32; i += 8)
    tile[ly + i][lx] = src[(size_t)(k0 + ly + i) * Nd + n0 + lx];
  __syncthreads();
#pragma unroll
  for (int i = 0; i < 32; i += 8)
    dst[(size_t)(n0 + ly + i) * Kd + k0 + lx] = f2bf(tile[lx][ly + i]);
}

__global__ __launch_bounds__(256) void zero_f32_kernel(float* p, int n) {
  int i = blockIdx.x * 256 + threadIdx.x;
  if (i < n) p[i] = 0.f;
}

// ---------------- QKV projection: [8192,512] x Wt[1536,512] ----------------
// WG=128 (4 waves), 128x64 tile; wave = 32 rows (2 A-tiles) x 64 cols.
// A 64-wide col tile maps to a single (q|k|v, head). V written transposed.
__global__ __launch_bounds__(128) void qkv_gemm(const __bf16* __restrict__ X,
                                                const __bf16* __restrict__ WT,
                                                __bf16* __restrict__ qb,
                                                __bf16* __restrict__ kb,
                                                __bf16* __restrict__ vbT) {
  __shared__ __bf16 At[128][40];
  __shared__ __bf16 BtT[64][40];
  __shared__ __bf16 Cbuf[128 * 72];  // q/k: [128][72]; v: transposed [64][136]
  const int tid = threadIdx.x, wave = tid >> 5, lane = tid & 31;
  const int m0 = blockIdx.y * 128, n0 = blockIdx.x * 64;
  v8f acc[2][4];
#pragma unroll
  for (int u = 0; u < 2; ++u)
#pragma unroll
    for (int t = 0; t < 4; ++t)
#pragma unroll
      for (int i = 0; i < 8; ++i) acc[u][t][i] = 0.f;

  for (int kk = 0; kk < CC; kk += 32) {
    {  // stage A: 128x32, one row per thread
      const __bf16* sa = X + (size_t)(m0 + tid) * CC + kk;
#pragma unroll
      for (int i = 0; i < 32; ++i) At[tid][i] = sa[i];
    }
    {  // stage Bt (transposed weights): 64 n-rows x 32 k
      int row = tid >> 1, seg = tid & 1;
      const __bf16* sb = WT + (size_t)(n0 + row) * CC + kk + seg * 16;
#pragma unroll
      for (int i = 0; i < 16; ++i) BtT[row][seg * 16 + i] = sb[i];
    }
    __syncthreads();
    v16bf a0 = load_a_bf(&At[0][0], 40, wave * 32, 0, lane);
    v16bf a1 = load_a_bf(&At[0][0], 40, wave * 32 + 16, 0, lane);
#pragma unroll
    for (int t = 0; t < 4; ++t) {
      v16bf b = load_bT_bf(&BtT[0][0], 40, t * 16, 0, lane);
      acc[0][t] = wmma_bf16(a0, b, acc[0][t]);
      acc[1][t] = wmma_bf16(a1, b, acc[1][t]);
    }
    __syncthreads();
  }
  const int t3 = n0 >> 9, h = (n0 & 511) >> 6;
  const int bidx = m0 >> 12, nbase = m0 & (NN - 1);
  const int half8 = (lane >> 4) << 3, col = lane & 15;
  if (t3 < 2) {
    // stage C tile [row][dd] then coalesced row writes; fold d^-0.5 into q
    const float sc = (t3 == 0) ? 0.125f : 1.0f;
#pragma unroll
    for (int u = 0; u < 2; ++u)
#pragma unroll
      for (int t = 0; t < 4; ++t)
#pragma unroll
        for (int r = 0; r < 8; ++r)
          Cbuf[(wave * 32 + u * 16 + r + half8) * 72 + t * 16 + col] =
              f2bf(acc[u][t][r] * sc);
    __syncthreads();
    __bf16* dstb = t3 ? kb : qb;
    __bf16* d = dstb + (((size_t)bidx * HH + h) * NN + nbase + tid) * DDIM;
#pragma unroll
    for (int i = 0; i < 64; ++i) d[i] = Cbuf[tid * 72 + i];
  } else {
    // transposed stage [dd][row]: 8 consecutive r are contiguous (16B stores)
#pragma unroll
    for (int u = 0; u < 2; ++u)
#pragma unroll
      for (int t = 0; t < 4; ++t) {
        int base = wave * 32 + u * 16 + half8;
#pragma unroll
        for (int r = 0; r < 8; ++r)
          Cbuf[(t * 16 + col) * 136 + base + r] = f2bf(acc[u][t][r]);
      }
    __syncthreads();
    int dd = tid >> 1, seg = tid & 1;
    __bf16* d = vbT + (((size_t)bidx * HH + h) * DDIM + dd) * NN + nbase + seg * 64;
#pragma unroll
    for (int i = 0; i < 64; ++i) d[i] = Cbuf[dd * 136 + seg * 64 + i];
  }
}

// ---------------- attention pass 1: row max m and sumexp l ----------------
// Row statistics reduced in registers via half-wave shuffles (no S in LDS).
__global__ __launch_bounds__(128) void attn_pass1(const __bf16* __restrict__ qb,
                                                  const __bf16* __restrict__ kb,
                                                  float* __restrict__ mbuf,
                                                  float* __restrict__ lbuf) {
  __shared__ __bf16 Kblk[64][72];
  const int tid = threadIdx.x, wave = tid >> 5, lane = tid & 31;
  const int b = blockIdx.z, h = blockIdx.y, bh = b * HH + h;
  const int q0 = blockIdx.x * 64;
  const __bf16* Q = qb + (size_t)bh * NN * DDIM;
  const __bf16* Kg = kb + (size_t)bh * NN * DDIM;
  v16bf aq[2];
#pragma unroll
  for (int s = 0; s < 2; ++s) aq[s] = load_a_bf(Q, DDIM, q0 + wave * 16, s * 32, lane);
  float m_run[8], l_run[8];
#pragma unroll
  for (int r = 0; r < 8; ++r) { m_run[r] = -__builtin_inff(); l_run[r] = 0.f; }

  for (int kbk = 0; kbk < NN / 64; ++kbk) {
    {
      int row = tid >> 1, seg = tid & 1;
      const __bf16* src = Kg + (size_t)(kbk * 64 + row) * DDIM + seg * 32;
#pragma unroll
      for (int i = 0; i < 32; ++i) Kblk[row][seg * 32 + i] = src[i];
    }
    __syncthreads();
    v8f c[4];
#pragma unroll
    for (int t = 0; t < 4; ++t) {
#pragma unroll
      for (int i = 0; i < 8; ++i) c[t][i] = 0.f;
#pragma unroll
      for (int s = 0; s < 2; ++s)
        c[t] = wmma_bf16(aq[s], load_bT_bf(&Kblk[0][0], 72, t * 16, s * 32, lane), c[t]);
    }
    // per-row online softmax stats; row r+8*(lane>>4) lives in 16-lane halves
#pragma unroll
    for (int r = 0; r < 8; ++r) {
      float mx = m_run[r];
#pragma unroll
      for (int t = 0; t < 4; ++t) mx = fmaxf(mx, c[t][r]);
#pragma unroll
      for (int msk = 1; msk < 16; msk <<= 1) mx = fmaxf(mx, __shfl_xor(mx, msk, 32));
      float sum = 0.f;
#pragma unroll
      for (int t = 0; t < 4; ++t) sum += __expf(c[t][r] - mx);
#pragma unroll
      for (int msk = 1; msk < 16; msk <<= 1) sum += __shfl_xor(sum, msk, 32);
      l_run[r] = l_run[r] * __expf(m_run[r] - mx) + sum;
      m_run[r] = mx;
    }
    __syncthreads();
  }
  if ((lane & 15) == 0) {
    size_t base = (size_t)bh * NN + q0 + wave * 16 + ((lane >> 4) << 3);
#pragma unroll
    for (int r = 0; r < 8; ++r) {
      mbuf[base + r] = m_run[r];
      lbuf[base + r] = l_run[r];
    }
  }
}

// ------- attention pass 2: O = P.V, colsum += per-key attention mass -------
__global__ __launch_bounds__(128) void attn_pass2(const __bf16* __restrict__ qb,
                                                  const __bf16* __restrict__ kb,
                                                  const __bf16* __restrict__ vbT,
                                                  const float* __restrict__ mbuf,
                                                  const float* __restrict__ lbuf,
                                                  float* __restrict__ colsum,
                                                  __bf16* __restrict__ outb) {
  __shared__ __bf16 Kblk[64][72];
  __shared__ __bf16 VblkT[64][72];  // [dd][key]
  __shared__ __bf16 Pl[64][72];
  __shared__ float colacc[64];
  const int tid = threadIdx.x, wave = tid >> 5, lane = tid & 31;
  const int b = blockIdx.z, h = blockIdx.y, bh = b * HH + h;
  const int q0 = blockIdx.x * 64;
  const __bf16* Q = qb + (size_t)bh * NN * DDIM;
  const __bf16* Kg = kb + (size_t)bh * NN * DDIM;
  const __bf16* VgT = vbT + (size_t)bh * DDIM * NN;  // [dd][NN]
  v16bf aq[2];
#pragma unroll
  for (int s = 0; s < 2; ++s) aq[s] = load_a_bf(Q, DDIM, q0 + wave * 16, s * 32, lane);
  float mv[8], iv[8];
  {
    size_t base = (size_t)bh * NN + q0 + wave * 16 + ((lane >> 4) << 3);
#pragma unroll
    for (int r = 0; r < 8; ++r) {
      mv[r] = mbuf[base + r];
      iv[r] = 1.0f / lbuf[base + r];
    }
  }
  v8f O[4];
#pragma unroll
  for (int t = 0; t < 4; ++t)
#pragma unroll
    for (int i = 0; i < 8; ++i) O[t][i] = 0.f;

  for (int kbk = 0; kbk < NN / 64; ++kbk) {
    {
      int row = tid >> 1, seg = tid & 1;
      const __bf16* sk = Kg + (size_t)(kbk * 64 + row) * DDIM + seg * 32;
      const __bf16* sv = VgT + (size_t)row * NN + kbk * 64 + seg * 32;  // row = dd
#pragma unroll
      for (int i = 0; i < 32; ++i) {
        Kblk[row][seg * 32 + i] = sk[i];
        VblkT[row][seg * 32 + i] = sv[i];
      }
    }
    if (tid < 64) colacc[tid] = 0.f;
    __syncthreads();
    v8f c[4];
#pragma unroll
    for (int t = 0; t < 4; ++t) {
#pragma unroll
      for (int i = 0; i < 8; ++i) c[t][i] = 0.f;
#pragma unroll
      for (int s = 0; s < 2; ++s)
        c[t] = wmma_bf16(aq[s], load_bT_bf(&Kblk[0][0], 72, t * 16, s * 32, lane), c[t]);
    }
    // P = exp(s - m)/l in registers; bf16 P tile to LDS; column partials in regs
#pragma unroll
    for (int t = 0; t < 4; ++t) {
      float cp = 0.f;
#pragma unroll
      for (int r = 0; r < 8; ++r) {
        float p = __expf(c[t][r] - mv[r]) * iv[r];
        c[t][r] = p;
        cp += p;
      }
      cp += __shfl_xor(cp, 16, 32);  // combine the two half-wave row groups
#pragma unroll
      for (int r = 0; r < 8; ++r)
        Pl[wave * 16 + r + ((lane >> 4) << 3)][t * 16 + (lane & 15)] = f2bf(c[t][r]);
      if (lane < 16) atomicAdd(&colacc[t * 16 + lane], cp);
    }
    __syncthreads();
    if (tid < 64)
      atomicAdd(&colsum[(size_t)b * NN + kbk * 64 + tid], colacc[tid]);
    // O += P(16x64) * V(64x64-slice); both operand frags contiguous LDS loads
#pragma unroll
    for (int s = 0; s < 2; ++s) {
      v16bf ap = load_a_bf(&Pl[0][0], 72, wave * 16, s * 32, lane);
#pragma unroll
      for (int t = 0; t < 4; ++t) {
        v16bf bvf = load_bT_bf(&VblkT[0][0], 72, t * 16, s * 32, lane);
        O[t] = wmma_bf16(ap, bvf, O[t]);
      }
    }
    __syncthreads();
  }
  // stage O (bf16) in LDS, then coalesced write to outb[b][q][h*64+dd]
#pragma unroll
  for (int t = 0; t < 4; ++t)
#pragma unroll
    for (int r = 0; r < 8; ++r)
      Pl[wave * 16 + r + ((lane >> 4) << 3)][t * 16 + (lane & 15)] = f2bf(O[t][r]);
  __syncthreads();
  {
    int row = tid >> 1, seg = tid & 1;
    __bf16* d = outb + ((size_t)b * NN + q0 + row) * CC + h * DDIM + seg * 32;
#pragma unroll
    for (int i = 0; i < 32; ++i) d[i] = Pl[row][seg * 32 + i];
  }
}

// ---------------- output projection: [8192,512] x Wt[512,512] + bias ----------------
__global__ __launch_bounds__(128) void proj_gemm(const __bf16* __restrict__ X,
                                                 const __bf16* __restrict__ WT,
                                                 const float* __restrict__ bias,
                                                 float* __restrict__ out) {
  __shared__ __bf16 At[128][40];
  __shared__ __bf16 BtT[64][40];
  __shared__ float Cf[128][65];
  const int tid = threadIdx.x, wave = tid >> 5, lane = tid & 31;
  const int m0 = blockIdx.y * 128, n0 = blockIdx.x * 64;
  v8f acc[2][4];
#pragma unroll
  for (int u = 0; u < 2; ++u)
#pragma unroll
    for (int t = 0; t < 4; ++t)
#pragma unroll
      for (int i = 0; i < 8; ++i) acc[u][t][i] = 0.f;

  for (int kk = 0; kk < CC; kk += 32) {
    {
      const __bf16* sa = X + (size_t)(m0 + tid) * CC + kk;
#pragma unroll
      for (int i = 0; i < 32; ++i) At[tid][i] = sa[i];
    }
    {
      int row = tid >> 1, seg = tid & 1;
      const __bf16* sb = WT + (size_t)(n0 + row) * CC + kk + seg * 16;
#pragma unroll
      for (int i = 0; i < 16; ++i) BtT[row][seg * 16 + i] = sb[i];
    }
    __syncthreads();
    v16bf a0 = load_a_bf(&At[0][0], 40, wave * 32, 0, lane);
    v16bf a1 = load_a_bf(&At[0][0], 40, wave * 32 + 16, 0, lane);
#pragma unroll
    for (int t = 0; t < 4; ++t) {
      v16bf b = load_bT_bf(&BtT[0][0], 40, t * 16, 0, lane);
      acc[0][t] = wmma_bf16(a0, b, acc[0][t]);
      acc[1][t] = wmma_bf16(a1, b, acc[1][t]);
    }
    __syncthreads();
  }
#pragma unroll
  for (int u = 0; u < 2; ++u)
#pragma unroll
    for (int t = 0; t < 4; ++t)
#pragma unroll
      for (int r = 0; r < 8; ++r)
        Cf[wave * 32 + u * 16 + r + ((lane >> 4) << 3)][t * 16 + (lane & 15)] =
            acc[u][t][r];
  __syncthreads();
  {
    float* d = out + (size_t)(m0 + tid) * CC + n0;
    const float* bs = bias + n0;
#pragma unroll
    for (int i = 0; i < 64; ++i) d[i] = Cf[tid][i] + bs[i];
  }
}

// ---------------- top-K selection (descending, ties -> smaller index) ----------------
__global__ __launch_bounds__(256) void topk_kernel(const float* __restrict__ colsum,
                                                   int* __restrict__ idx) {
  __shared__ float imp[NN];
  __shared__ float rv[256];
  __shared__ int ri[256];
  const int b = blockIdx.x, tid = threadIdx.x;
  for (int i = tid; i < NN; i += 256) imp[i] = colsum[(size_t)b * NN + i];
  __syncthreads();
  for (int r = 0; r < KP; ++r) {
    float bv = -__builtin_inff();
    int bi = NN;
    for (int i = tid; i < NN; i += 256) {
      float v = imp[i];
      if (v > bv || (v == bv && i < bi)) { bv = v; bi = i; }
    }
    rv[tid] = bv;
    ri[tid] = bi;
    __syncthreads();
    for (int off = 128; off > 0; off >>= 1) {
      if (tid < off) {
        float ov = rv[tid + off];
        int oi = ri[tid + off];
        if (ov > rv[tid] || (ov == rv[tid] && oi < ri[tid])) { rv[tid] = ov; ri[tid] = oi; }
      }
      __syncthreads();
    }
    if (tid == 0) {
      idx[(size_t)b * KP + r] = ri[0];
      imp[ri[0]] = -__builtin_inff();
    }
    __syncthreads();
  }
}

__global__ __launch_bounds__(128) void gather_kernel(const float* __restrict__ proj,
                                                     const int* __restrict__ idx,
                                                     float* __restrict__ out) {
  const int b = blockIdx.y, r = blockIdx.x;
  const int row = idx[(size_t)b * KP + r];
  const float* src = proj + ((size_t)b * NN + row) * CC;
  float* dst = out + ((size_t)b * KP + r) * CC;
  for (int c = threadIdx.x; c < CC; c += 128) dst[c] = src[c];
}

extern "C" void kernel_launch(void* const* d_in, const int* in_sizes, int n_in,
                              void* d_out, int out_size, void* d_ws, size_t ws_size,
                              hipStream_t stream) {
  const float* x = (const float*)d_in[0];
  const float* w_qkv = (const float*)d_in[1];
  const float* w_proj = (const float*)d_in[2];
  const float* b_proj = (const float*)d_in[3];
  float* out = (float*)d_out;

  char* ws = (char*)d_ws;
  size_t off = 0;
  auto alloc = [&](size_t bytes) -> char* {
    char* p = ws + off;
    off = (off + bytes + 255) & ~(size_t)255;
    return p;
  };
  __bf16* xb = (__bf16*)alloc((size_t)BB * NN * CC * 2);
  __bf16* wqkvT = (__bf16*)alloc((size_t)CC * C3 * 2);   // [1536][512]
  __bf16* wprojT = (__bf16*)alloc((size_t)CC * CC * 2);  // [512][512]
  __bf16* qb = (__bf16*)alloc((size_t)BB * HH * NN * DDIM * 2);
  __bf16* kb = (__bf16*)alloc((size_t)BB * HH * NN * DDIM * 2);
  __bf16* vbT = (__bf16*)alloc((size_t)BB * HH * DDIM * NN * 2);  // [b][h][dd][n]
  float* mbuf = (float*)alloc((size_t)BB * HH * NN * 4);
  float* lbuf = (float*)alloc((size_t)BB * HH * NN * 4);
  __bf16* outb = (__bf16*)alloc((size_t)BB * NN * CC * 2);
  float* colsum = (float*)alloc((size_t)BB * NN * 4);
  float* proj = (float*)alloc((size_t)BB * NN * CC * 4);
  int* idx = (int*)alloc((size_t)BB * KP * 4);
  (void)ws_size; (void)in_sizes; (void)n_in; (void)out_size;

  // 1) casts to bf16 (weights transposed so GEMM B-frags are contiguous)
  {
    int n = BB * NN * CC;
    cast_bf16_kernel<<<(n + 255) / 256, 256, 0, stream>>>(x, xb, n);
    transpose_cast_kernel<<<dim3(C3 / 32, CC / 32), 256, 0, stream>>>(w_qkv, wqkvT,
                                                                      CC, C3);
    transpose_cast_kernel<<<dim3(CC / 32, CC / 32), 256, 0, stream>>>(w_proj, wprojT,
                                                                      CC, CC);
  }
  // 2) QKV projection (V stored transposed)
  qkv_gemm<<<dim3(C3 / 64, (BB * NN) / 128), 128, 0, stream>>>(xb, wqkvT, qb, kb, vbT);
  // 3) attention pass 1 (m, l)
  attn_pass1<<<dim3(NN / 64, HH, BB), 128, 0, stream>>>(qb, kb, mbuf, lbuf);
  // 4) zero colsum, then pass 2 (O + column mass)
  zero_f32_kernel<<<(BB * NN + 255) / 256, 256, 0, stream>>>(colsum, BB * NN);
  attn_pass2<<<dim3(NN / 64, HH, BB), 128, 0, stream>>>(qb, kb, vbT, mbuf, lbuf,
                                                        colsum, outb);
  // 5) output projection (+bias)
  proj_gemm<<<dim3(CC / 64, (BB * NN) / 128), 128, 0, stream>>>(outb, wprojT, b_proj,
                                                                proj);
  // 6) top-K pivotal tokens, 7) gather
  topk_kernel<<<BB, 256, 0, stream>>>(colsum, idx);
  gather_kernel<<<dim3(KP, BB), 128, 0, stream>>>(proj, idx, out);
}
